// SSMClassifier_8512625181099
// MI455X (gfx1250) — matrix operand order repaired and checked by reading, not compiled, not run on hardware
//
#include <hip/hip_runtime.h>
#include <hip/hip_bf16.h>
#include <math.h>

// ---------------------------------------------------------------------------
// SSM classifier for MI455X (gfx1250, wave32, WMMA bf16 16x16x32).
// Chunked linear scan: s_t = s_{t-1} A^T + x_t M^T + mb,  y = S C^T, GELU+LN.
// ---------------------------------------------------------------------------

typedef __bf16 bf16;
typedef __attribute__((ext_vector_type(16))) __bf16 v16bf;
typedef __attribute__((ext_vector_type(8)))  float  v8f;

#define DMODEL 1024
#define DSTATE 256
#define INDIM  28
#define NBATCH 64
#define SEQLEN 1024
#define CK     32      // chunk length
#define NC     32      // SEQLEN / CK
#define NBB    4       // NBATCH / 16

// ---------------- WMMA fragment helpers (wave32 layouts per CDNA5 ISA) ------

// A-operand 16x32 bf16 from row-major source (row stride ld elements).
// lane m=lane&15, half=lane>>4: elems 0..7 = K[half*8 .. +7], 8..15 = K[16+half*8 .. +7]
static __device__ inline v16bf ldA(const bf16* base, int ld, int lane) {
  int m = lane & 15, h = lane >> 4;
  const bf16* p = base + m * ld + h * 8;
  union { v16bf v; uint4 q[2]; } u;
  u.q[0] = *(const uint4*)p;
  u.q[1] = *(const uint4*)(p + 16);
  return u.v;
}

// A-operand from f32 source (LDS state), converting to bf16.
static __device__ inline v16bf ldA_f32(const float* base, int ld, int lane) {
  int m = lane & 15, h = lane >> 4;
  const float* p = base + m * ld;
  v16bf a;
#pragma unroll
  for (int i = 0; i < 8; ++i) a[i] = (__bf16)p[h * 8 + i];
#pragma unroll
  for (int i = 0; i < 8; ++i) a[8 + i] = (__bf16)p[16 + h * 8 + i];
  return a;
}

// B-operand 32x16 bf16. Source is "n-major": src[n*ld + k] == B(k,n).
// lane n=lane&15, half=lane>>4: elems 0..7 = K[half*8 .. +7], 8..15 = K[16+half*8 .. +7]
static __device__ inline v16bf ldB(const bf16* base, int ld, int lane) {
  int n = lane & 15, h = lane >> 4;
  const bf16* p = base + n * ld + h * 8;
  union { v16bf v; uint4 q[2]; } u;
  u.q[0] = *(const uint4*)p;
  u.q[1] = *(const uint4*)(p + 16);
  return u.v;
}

// C/D accumulator init from bf16 row-major [16][ld] tile (col base pre-added).
static __device__ inline v8f ldC_bf16(const bf16* base, int ld, int lane) {
  int n = lane & 15, h = lane >> 4;
  v8f c;
#pragma unroll
  for (int v = 0; v < 8; ++v) c[v] = (float)base[(v + 8 * h) * ld + n];
  return c;
}

// Store D (8 f32 per lane) to row-major f32 [16][ld] tile.
static __device__ inline void stC_f32(float* base, int ld, int lane, v8f a) {
  int n = lane & 15, h = lane >> 4;
#pragma unroll
  for (int v = 0; v < 8; ++v) base[(v + 8 * h) * ld + n] = a[v];
}

static __device__ inline v8f wmma_bf(v16bf a, v16bf b, v8f c) {
  return __builtin_amdgcn_wmma_f32_16x16x32_bf16(false, a, false, b, (short)0, c,
                                                 false, false);
}

// Cooperative 16x256 bf16 stage (256 threads, 32B per thread, coalesced).
static __device__ inline void stage16x256(bf16* dst, const bf16* src,
                                          size_t srcRowStride, int tid) {
  int r = tid >> 4, seg = (tid & 15) << 4;
  const bf16* s = src + (size_t)r * srcRowStride + seg;
  bf16* d = dst + r * 256 + seg;
  *(uint4*)d       = *(const uint4*)s;
  *((uint4*)d + 1) = *((const uint4*)s + 1);
}

// Cooperative f32 LDS [16][256] -> bf16 global (row stride), coalesced.
static __device__ inline void store16x256_bf16(bf16* dst, size_t dstRowStride,
                                               const float* src, int tid) {
  int r = tid >> 4, seg = (tid & 15) << 4;
  const float* s = src + r * 256 + seg;
  bf16* d = dst + (size_t)r * dstRowStride + seg;
  union { uint4 q[2]; bf16 h[16]; } u;
#pragma unroll
  for (int i = 0; i < 16; ++i) u.h[i] = (__bf16)s[i];
  *(uint4*)d       = u.q[0];
  *((uint4*)d + 1) = u.q[1];
}

static __device__ inline float fixnum(float v) {
  if (isnan(v)) return 0.0f;
  if (isinf(v)) return v > 0.0f ? 1000000.0f : -1000000.0f;
  return v;
}

// ---------------- small prep kernels ---------------------------------------

// M = Bm @ W_in (256x28), mb = Bm @ b_in (256)
__global__ void k_projM(const float* __restrict__ Bm, const float* __restrict__ Win,
                        const float* __restrict__ bin, float* __restrict__ M,
                        float* __restrict__ mb) {
  int idx = blockIdx.x * 256 + threadIdx.x;
  if (idx < 256 * INDIM) {
    int n = idx / INDIM, i = idx % INDIM;
    float s = 0.f;
    for (int d = 0; d < DMODEL; ++d) s += Bm[n * DMODEL + d] * Win[d * INDIM + i];
    M[idx] = s;
  } else if (idx < 256 * INDIM + 256) {
    int n = idx - 256 * INDIM;
    float s = 0.f;
    for (int d = 0; d < DMODEL; ++d) s += Bm[n * DMODEL + d] * bin[d];
    mb[n] = s;
  }
}

// A -> Apow[0]=A^1 (bf16), At = A^T (bf16, B-source for power GEMMs), C -> bf16
__global__ void k_cvt(const float* __restrict__ A, const float* __restrict__ C,
                      bf16* __restrict__ A1, bf16* __restrict__ At,
                      bf16* __restrict__ Cb) {
  int idx = blockIdx.x * 256 + threadIdx.x;
  if (idx < 65536) {
    A1[idx] = (__bf16)A[idx];
  } else if (idx < 131072) {
    int i = idx - 65536;
    int r = i >> 8, c = i & 255;
    At[i] = (__bf16)A[c * 256 + r];
  } else {
    int i = idx - 131072;
    Cb[i] = (__bf16)C[i];
  }
}

// U[b,t,:] = x[b,t,:] @ M^T + mb  (bf16 out)
__global__ void k_upro(const float* __restrict__ x, const float* __restrict__ M,
                       const float* __restrict__ mb, bf16* __restrict__ U) {
  __shared__ float xl[INDIM];
  int bt = blockIdx.x, n = threadIdx.x;
  if (n < INDIM) xl[n] = x[(size_t)bt * INDIM + n];
  __syncthreads();
  float s = mb[n];
#pragma unroll
  for (int i = 0; i < INDIM; ++i) s += xl[i] * M[n * INDIM + i];
  U[(size_t)bt * DSTATE + n] = (__bf16)s;
}

// Aout = Aprev @ A  (256x256 bf16 GEMM via WMMA; B-source = At n-major)
__global__ void k_power(const bf16* __restrict__ Aprev, const bf16* __restrict__ At,
                        bf16* __restrict__ Aout) {
  int tid = threadIdx.x, lane = tid & 31, wave = tid >> 5;
  int rb = blockIdx.x;  // 16 row-blocks of 16
  const bf16* Abase = Aprev + (size_t)rb * 16 * 256;
  v8f a0 = {0, 0, 0, 0, 0, 0, 0, 0}, a1 = a0;
  for (int kk = 0; kk < 8; ++kk) {
    v16bf af = ldA(Abase + kk * 32, 256, lane);
    v16bf b0 = ldB(At + (size_t)(wave * 2 + 0) * 16 * 256 + kk * 32, 256, lane);
    v16bf b1 = ldB(At + (size_t)(wave * 2 + 1) * 16 * 256 + kk * 32, 256, lane);
    a0 = wmma_bf(af, b0, a0);
    a1 = wmma_bf(af, b1, a1);
  }
  int n = lane & 15, h = lane >> 4;
#pragma unroll
  for (int v = 0; v < 8; ++v) {
    size_t row = (size_t)(rb * 16 + v + 8 * h) * 256;
    Aout[row + (wave * 2 + 0) * 16 + n] = (__bf16)a0[v];
    Aout[row + (wave * 2 + 1) * 16 + n] = (__bf16)a1[v];
  }
}

// ---------------- phase 1: local scans (zero-init, 32 steps) ----------------
// grid: NC*NBB workgroups; each: 16 batch rows, state (16x256) in LDS (f32).
__global__ void k_scan(const bf16* __restrict__ U, const bf16* __restrict__ A1,
                       bf16* __restrict__ S) {
  __shared__ float sbuf[16 * 256];
  __shared__ __align__(16) bf16 ubuf[16 * 256];
  int tid = threadIdx.x, lane = tid & 31, wave = tid >> 5;
  int c = blockIdx.x & (NC - 1);
  int bb = blockIdx.x >> 5;
  for (int i = tid; i < 16 * 256; i += 256) sbuf[i] = 0.f;
  const size_t rss = (size_t)SEQLEN * DSTATE;
  for (int j = 0; j < CK; ++j) {
    int t = c * CK + j;
    stage16x256(ubuf, U + ((size_t)(bb * 16) * SEQLEN + t) * DSTATE, rss, tid);
    __syncthreads();
    v8f a0 = ldC_bf16(ubuf + (wave * 2 + 0) * 16, 256, lane);
    v8f a1 = ldC_bf16(ubuf + (wave * 2 + 1) * 16, 256, lane);
    for (int kk = 0; kk < 8; ++kk) {
      v16bf af = ldA_f32(sbuf + kk * 32, 256, lane);
      v16bf b0 = ldB(A1 + (size_t)(wave * 2 + 0) * 16 * 256 + kk * 32, 256, lane);
      v16bf b1 = ldB(A1 + (size_t)(wave * 2 + 1) * 16 * 256 + kk * 32, 256, lane);
      a0 = wmma_bf(af, b0, a0);
      a1 = wmma_bf(af, b1, a1);
    }
    __syncthreads();
    stC_f32(sbuf + (wave * 2 + 0) * 16, 256, lane, a0);
    stC_f32(sbuf + (wave * 2 + 1) * 16, 256, lane, a1);
    __syncthreads();
    store16x256_bf16(S + ((size_t)(bb * 16) * SEQLEN + t) * DSTATE, rss, sbuf, tid);
  }
}

// ---------------- phase 2: carry propagation over chunks --------------------
// E[bb, c] = state entering chunk c.  E_{c+1} = l_end(c) + E_c @ (A^T)^CK.
__global__ void k_carry(const bf16* __restrict__ S, const bf16* __restrict__ A32,
                        bf16* __restrict__ E) {
  __shared__ float ebuf[16 * 256];
  __shared__ __align__(16) bf16 ubuf[16 * 256];
  int tid = threadIdx.x, lane = tid & 31, wave = tid >> 5;
  int bb = blockIdx.x;
  for (int i = tid; i < 16 * 256; i += 256) ebuf[i] = 0.f;
  __syncthreads();
  const size_t rss = (size_t)SEQLEN * DSTATE;
  for (int c = 0; c < NC; ++c) {
    store16x256_bf16(E + (size_t)(bb * NC + c) * 16 * 256, 256, ebuf, tid);
    if (c == NC - 1) break;
    int t = c * CK + (CK - 1);
    stage16x256(ubuf, S + ((size_t)(bb * 16) * SEQLEN + t) * DSTATE, rss, tid);
    __syncthreads();
    v8f a0 = ldC_bf16(ubuf + (wave * 2 + 0) * 16, 256, lane);
    v8f a1 = ldC_bf16(ubuf + (wave * 2 + 1) * 16, 256, lane);
    for (int kk = 0; kk < 8; ++kk) {
      v16bf af = ldA_f32(ebuf + kk * 32, 256, lane);
      v16bf b0 = ldB(A32 + (size_t)(wave * 2 + 0) * 16 * 256 + kk * 32, 256, lane);
      v16bf b1 = ldB(A32 + (size_t)(wave * 2 + 1) * 16 * 256 + kk * 32, 256, lane);
      a0 = wmma_bf(af, b0, a0);
      a1 = wmma_bf(af, b1, a1);
    }
    __syncthreads();
    stC_f32(ebuf + (wave * 2 + 0) * 16, 256, lane, a0);
    stC_f32(ebuf + (wave * 2 + 1) * 16, 256, lane, a1);
    __syncthreads();
  }
}

// ---------------- phase 3: apply carries (parallel) -------------------------
// s[b, cK+j] = l[b, cK+j] + E[b,c] @ (A^T)^(j+1);  (A^T)^m as B-op == A^m rows.
__global__ void k_apply(const bf16* __restrict__ E, const bf16* __restrict__ Apow,
                        bf16* __restrict__ S) {
  __shared__ __align__(16) bf16 lbuf[16 * 256];
  __shared__ __align__(16) bf16 ebuf[16 * 256];
  __shared__ float obuf[16 * 256];
  int tid = threadIdx.x, lane = tid & 31, wave = tid >> 5;
  int bx = blockIdx.x;
  int bb = bx & 3;
  int j = (bx >> 2) & 31;
  int c = 1 + (bx >> 7);
  int t = c * CK + j;
  const size_t rss = (size_t)SEQLEN * DSTATE;
  stage16x256(lbuf, S + ((size_t)(bb * 16) * SEQLEN + t) * DSTATE, rss, tid);
  stage16x256(ebuf, E + (size_t)(bb * NC + c) * 16 * 256, 256, tid);
  __syncthreads();
  const bf16* P = Apow + (size_t)j * 65536;  // A^(j+1)
  v8f a0 = ldC_bf16(lbuf + (wave * 2 + 0) * 16, 256, lane);
  v8f a1 = ldC_bf16(lbuf + (wave * 2 + 1) * 16, 256, lane);
  for (int kk = 0; kk < 8; ++kk) {
    v16bf af = ldA(ebuf + kk * 32, 256, lane);
    v16bf b0 = ldB(P + (size_t)(wave * 2 + 0) * 16 * 256 + kk * 32, 256, lane);
    v16bf b1 = ldB(P + (size_t)(wave * 2 + 1) * 16 * 256 + kk * 32, 256, lane);
    a0 = wmma_bf(af, b0, a0);
    a1 = wmma_bf(af, b1, a1);
  }
  stC_f32(obuf + (wave * 2 + 0) * 16, 256, lane, a0);
  stC_f32(obuf + (wave * 2 + 1) * 16, 256, lane, a1);
  __syncthreads();
  store16x256_bf16(S + ((size_t)(bb * 16) * SEQLEN + t) * DSTATE, rss, obuf, tid);
}

// ---------------- phase 4: y = S @ C^T, GELU, LayerNorm, store activations --
// One WG per 16 (b,t) rows; 8 waves x 8 N-tiles each; accumulators in regs.
__global__ void k_output(const bf16* __restrict__ S, const bf16* __restrict__ Cb,
                         const float* __restrict__ gamma, const float* __restrict__ beta,
                         float* __restrict__ act) {
  __shared__ __align__(16) bf16 sstage[16 * 256];
  __shared__ float gl[DMODEL], bl[DMODEL];
  __shared__ float reds[16][128];
  __shared__ float redq[16][128];
  __shared__ float mu_s[16], rs_s[16];
  int tid = threadIdx.x, lane = tid & 31, wave = tid >> 5;
  int b = blockIdx.x >> 6;
  int t0 = (blockIdx.x & 63) << 4;
  stage16x256(sstage, S + ((size_t)b * SEQLEN + t0) * DSTATE, 256, tid);
  for (int i = tid; i < DMODEL; i += 256) { gl[i] = gamma[i]; bl[i] = beta[i]; }
  __syncthreads();

  v8f acc[8];
#pragma unroll
  for (int i = 0; i < 8; ++i) acc[i] = (v8f){0, 0, 0, 0, 0, 0, 0, 0};
  for (int kk = 0; kk < 8; ++kk) {
    v16bf af = ldA(sstage + kk * 32, 256, lane);
#pragma unroll
    for (int i = 0; i < 8; ++i) {
      int nt = wave * 8 + i;
      v16bf bf = ldB(Cb + (size_t)nt * 16 * 256 + kk * 32, 256, lane);
      acc[i] = wmma_bf(af, bf, acc[i]);
    }
  }

  int h = lane >> 4, n = lane & 15;
  float ps[8], pq[8];
#pragma unroll
  for (int v = 0; v < 8; ++v) { ps[v] = 0.f; pq[v] = 0.f; }
#pragma unroll
  for (int i = 0; i < 8; ++i) {
#pragma unroll
    for (int v = 0; v < 8; ++v) {
      float x = acc[i][v];
      float g = 0.5f * x * (1.0f + erff(x * 0.70710678118654752f));  // exact GELU
      acc[i][v] = g;
      ps[v] += g;
      pq[v] += g * g;
    }
  }
  int col = wave * 16 + n;
#pragma unroll
  for (int v = 0; v < 8; ++v) {
    reds[v + 8 * h][col] = ps[v];
    redq[v + 8 * h][col] = pq[v];
  }
  __syncthreads();
  if (tid < 16) {
    float s = 0.f, q = 0.f;
    for (int i = 0; i < 128; ++i) { s += reds[tid][i]; q += redq[tid][i]; }
    float mu = s * (1.0f / DMODEL);
    float var = q * (1.0f / DMODEL) - mu * mu;
    mu_s[tid] = mu;
    rs_s[tid] = rsqrtf(var + 1e-5f);
  }
  __syncthreads();
#pragma unroll
  for (int v = 0; v < 8; ++v) {
    int m = v + 8 * h;
    float mu = mu_s[m], rs = rs_s[m];
    size_t rowbase = ((size_t)b * SEQLEN + (t0 + m)) * (size_t)DMODEL;
#pragma unroll
    for (int i = 0; i < 8; ++i) {
      int d = (wave * 8 + i) * 16 + n;
      float o = (acc[i][v] - mu) * rs * gl[d] + bl[d];
      act[rowbase + d] = fixnum(o);
    }
  }
}

// ---------------- pooling + FC ---------------------------------------------
__global__ void k_pool(const float* __restrict__ act, float* __restrict__ pooled) {
  int b = blockIdx.x >> 2, d = ((blockIdx.x & 3) << 8) + threadIdx.x;
  const float* p = act + (size_t)b * SEQLEN * DMODEL + d;
  float s = 0.f;
  for (int t = 0; t < SEQLEN; ++t) s += p[(size_t)t * DMODEL];
  pooled[b * DMODEL + d] = s * (1.0f / SEQLEN);
}

__global__ void k_fc(const float* __restrict__ pooled, const float* __restrict__ Wfc,
                     const float* __restrict__ bfc, float* __restrict__ logits) {
  int idx = blockIdx.x * 256 + threadIdx.x;
  if (idx >= NBATCH * 10) return;
  int b = idx / 10, cl = idx % 10;
  float s = bfc[cl];
  const float* p = pooled + b * DMODEL;
  const float* w = Wfc + cl * DMODEL;
  for (int i = 0; i < DMODEL; ++i) s += p[i] * w[i];
  logits[idx] = s;
}

// ---------------- host driver -----------------------------------------------
extern "C" void kernel_launch(void* const* d_in, const int* in_sizes, int n_in,
                              void* d_out, int out_size, void* d_ws, size_t ws_size,
                              hipStream_t stream) {
  const float* x    = (const float*)d_in[0];
  const float* Win  = (const float*)d_in[1];
  const float* bin  = (const float*)d_in[2];
  const float* A    = (const float*)d_in[3];
  const float* Bm   = (const float*)d_in[4];
  const float* C    = (const float*)d_in[5];
  const float* gamma= (const float*)d_in[6];
  const float* beta = (const float*)d_in[7];
  const float* Wfc  = (const float*)d_in[8];
  const float* bfc  = (const float*)d_in[9];

  float* logits = (float*)d_out;
  float* act    = (float*)d_out + NBATCH * 10;

  char* ws = (char*)d_ws;
  size_t off = 0;
  bf16* S    = (bf16*)(ws + off); off += (size_t)NBATCH * SEQLEN * DSTATE * 2;   // 32 MB
  bf16* U    = (bf16*)(ws + off); off += (size_t)NBATCH * SEQLEN * DSTATE * 2;   // 32 MB
  bf16* Apow = (bf16*)(ws + off); off += (size_t)CK * 65536 * 2;                 // 4 MB
  bf16* At   = (bf16*)(ws + off); off += 65536 * 2;
  bf16* Cb   = (bf16*)(ws + off); off += (size_t)DMODEL * DSTATE * 2;
  float* M   = (float*)(ws + off); off += 256 * INDIM * 4 + 64;
  float* mb  = (float*)(ws + off); off += 256 * 4 + 64;
  bf16* E    = (bf16*)(ws + off); off += (size_t)NBB * NC * 16 * 256 * 2;
  float* pooled = (float*)(ws + off); off += (size_t)NBATCH * DMODEL * 4;
  (void)ws_size; (void)in_sizes; (void)n_in; (void)out_size;

  // prep: fold Bm@W_in, convert A/C to bf16, project inputs
  k_projM<<<29, 256, 0, stream>>>(Bm, Win, bin, M, mb);
  k_cvt<<<1536, 256, 0, stream>>>(A, C, Apow /*A^1*/, At, Cb);
  k_upro<<<NBATCH * SEQLEN, 256, 0, stream>>>(x, M, mb, U);

  // powers A^2..A^CK (each launch = global sync)
  for (int m = 2; m <= CK; ++m)
    k_power<<<16, 256, 0, stream>>>(Apow + (size_t)(m - 2) * 65536, At,
                                    Apow + (size_t)(m - 1) * 65536);

  // chunked scan
  k_scan<<<NC * NBB, 256, 0, stream>>>(U, Apow, S);
  k_carry<<<NBB, 256, 0, stream>>>(S, Apow + (size_t)(CK - 1) * 65536, E);
  k_apply<<<(NC - 1) * CK * NBB, 256, 0, stream>>>(E, Apow, S);

  // fused output GEMM + GELU + LayerNorm + nan_to_num
  k_output<<<NBATCH * (SEQLEN / 16), 256, 0, stream>>>(S, Cb, gamma, beta, act);

  // deterministic pooling + classifier head
  k_pool<<<NBATCH * 4, 256, 0, stream>>>(act, pooled);
  k_fc<<<3, 256, 0, stream>>>(pooled, Wfc, bfc, logits);
}
